// Image_Encoder_41841571398449
// MI455X (gfx1250) — matrix operand order repaired and checked by reading, not compiled, not run on hardware
//
#include <hip/hip_runtime.h>
#include <hip/hip_bf16.h>

typedef __attribute__((ext_vector_type(16))) __bf16 v16bf;
typedef __attribute__((ext_vector_type(8)))  __bf16 bf16x8;
typedef __attribute__((ext_vector_type(8)))  float  v8f;

union ABFrag { v16bf v; bf16x8 h[2]; };

#if defined(__gfx1250__) && __has_builtin(__builtin_amdgcn_global_load_async_to_lds_b128) && \
    __has_builtin(__builtin_amdgcn_s_wait_asynccnt)
#define IE_HAVE_ASYNC_LDS 1
typedef int ie_v4i __attribute__((vector_size(16)));            // b128 payload type
typedef __attribute__((address_space(1))) ie_v4i as1_v4i;       // global
typedef __attribute__((address_space(3))) ie_v4i as3_v4i;       // LDS
#else
#define IE_HAVE_ASYNC_LDS 0
#endif

// ---------------------------------------------------------------------------
// Workspace layout (bytes). Total ~143 MB.
// ---------------------------------------------------------------------------
static const size_t OFF_ACT0  = 0;                 // bf16 NHWC ping (64 MB: layer0 input 32*256*256*16*2)
static const size_t OFF_ACT1  = 67108864;          // bf16 NHWC pong (<= 20 MB)
static const size_t OFF_CONV  = 88080384;          // fp32 [M][co] conv+bias out (<= 32 MB)
static const size_t OFF_WPREP = 121634816;         // bf16 [K][co] packed weights (<= 2 MB)
static const size_t OFF_STATS = 123731968;         // fp32 2*co scale/shift
static const size_t OFF_FEAT0 = 123736064;         // fp32 NHWC features (layers 5,8,11,12)
static const size_t OFF_FEAT1 = 137043968;
static const size_t OFF_FEAT2 = 141762560;
static const size_t OFF_FEAT3 = 142565376;

// ---------------------------------------------------------------------------
// Kernel 1: concat img_occ/img_unocc (NCHW fp32) -> NHWC bf16, channels padded to 16
// ---------------------------------------------------------------------------
__global__ void ie_concat_pad(const float* __restrict__ occ,
                              const float* __restrict__ unocc,
                              __bf16* __restrict__ act)
{
    int i = blockIdx.x * blockDim.x + threadIdx.x;       // pixel over B*256*256
    if (i >= 32 * 256 * 256) return;
    int b  = i >> 16;
    int yx = i & 65535;
    const float* o = occ   + (size_t)b * 3 * 65536 + yx;
    const float* u = unocc + (size_t)b * 3 * 65536 + yx;
    __bf16* p = act + (size_t)i * 16;
    p[0] = (__bf16)o[0]; p[1] = (__bf16)o[65536]; p[2] = (__bf16)o[131072];
    p[3] = (__bf16)u[0]; p[4] = (__bf16)u[65536]; p[5] = (__bf16)u[131072];
    #pragma unroll
    for (int c = 6; c < 16; ++c) p[c] = (__bf16)0.0f;
}

// ---------------------------------------------------------------------------
// Kernel 2: pack weights OIHW fp32 -> [K][co] bf16, K = pos*cpad + c, zero-padded
// ---------------------------------------------------------------------------
__global__ void ie_prep_weights(const float* __restrict__ w,
                                __bf16* __restrict__ wp,
                                int ci, int co, int lgc, int Ktot)
{
    int i = blockIdx.x * blockDim.x + threadIdx.x;       // over Ktot*co
    if (i >= Ktot * co) return;
    int K = i / co;
    int n = i - K * co;
    int pos = K >> lgc;
    int c   = K & ((1 << lgc) - 1);
    float v = (pos < 25 && c < ci) ? w[((size_t)n * ci + c) * 25 + pos] : 0.0f;
    wp[(size_t)K * co + n] = (__bf16)v;
}

// ---------------------------------------------------------------------------
// Kernel 3: implicit-GEMM conv via V_WMMA_F32_16X16X32_BF16.
// Templated on CPAD/STRIDE/CO so all K-decomposition math constant-folds.
// 4 waves/block; each wave computes a 32(M)x16(N) C block (2 accumulators
// sharing one B fragment -> 2 WMMAs per K-chunk). Weights staged in LDS in
// 32 KB K-windows via GLOBAL_LOAD_ASYNC_TO_LDS_B128 (ASYNCcnt) when the
// toolchain exposes it; next window prefetched with global_prefetch.
// ---------------------------------------------------------------------------
template <int CPAD, int STRIDE, int CO>
__global__ void __launch_bounds__(128)
ie_conv_wmma(const __bf16* __restrict__ act,   // NHWC bf16, [b][iy][ix][CPAD]
             const __bf16* __restrict__ wp,    // [K][CO] bf16
             const float*  __restrict__ bias,
             float*        __restrict__ out,   // [M][CO] fp32
             int Mtot, int Hin, int Hout)
{
    constexpr int NCHUNK  = (25 * CPAD + 31) / 32;
    constexpr int KTOT    = NCHUNK * 32;
    constexpr int NGROUPS = (NCHUNK + 31) / 32;

    __shared__ __align__(16) __bf16 lw[1024 * 16];       // 32 KB K-window of 16 N columns

    const int tn   = blockIdx.y;
    const int wave = threadIdx.x >> 5;
    const int lane = threadIdx.x & 31;
    const int half = lane >> 4;
    const int l16  = lane & 15;
    const int tm0  = (blockIdx.x * 4 + wave) * 2;        // two M tiles per wave

    // Decompose this lane's two A rows (M and M+16) once.
    const int HWo = Hout * Hout;
    int b0, oy0, ox0, b1, oy1, ox1;
    bool mv0, mv1;
    {
        int m0 = tm0 * 16 + l16;
        int m1 = m0 + 16;
        mv0 = (m0 < Mtot); mv1 = (m1 < Mtot);
        int mm0 = mv0 ? m0 : 0, mm1 = mv1 ? m1 : 0;
        b0 = mm0 / HWo; int r0 = mm0 - b0 * HWo; oy0 = r0 / Hout; ox0 = r0 - oy0 * Hout;
        b1 = mm1 / HWo; int r1 = mm1 - b1 * HWo; oy1 = r1 / Hout; ox1 = r1 - oy1 * Hout;
    }

    v8f acc0, acc1;
    #pragma unroll
    for (int r = 0; r < 8; ++r) { acc0[r] = 0.0f; acc1[r] = 0.0f; }

    #pragma unroll
    for (int g = 0; g < NGROUPS; ++g) {
        const int kbase  = g * 1024;
        const int kcount = (KTOT - kbase < 1024) ? (KTOT - kbase) : 1024;
        __syncthreads();
        // Stage [kcount][16] weight window in LDS: 2x 16-byte transfers per K row.
        {
            const int cnt = kcount * 2;
            for (int t = threadIdx.x; t < cnt; t += blockDim.x) {
                int Kl = t >> 1, part = (t & 1) * 8;
                const __bf16* src = wp + (size_t)(kbase + Kl) * CO + tn * 16 + part;
                __bf16*       dst = lw + Kl * 16 + part;
#if IE_HAVE_ASYNC_LDS
                __builtin_amdgcn_global_load_async_to_lds_b128(
                    (as1_v4i*)(__bf16*)src, (as3_v4i*)dst, 0, 0);
#else
                *reinterpret_cast<uint4*>(dst) = *reinterpret_cast<const uint4*>(src);
#endif
            }
#if IE_HAVE_ASYNC_LDS
            __builtin_amdgcn_s_wait_asynccnt(0);
#endif
        }
        // Prefetch next K-window into cache while this one is consumed.
        if (g + 1 < NGROUPS)
            __builtin_prefetch(wp + (size_t)(kbase + 1024) * CO + tn * 16 + (threadIdx.x & 63) * 8,
                               0, 0);
        __syncthreads();

        const int cend = ((NCHUNK - g * 32) < 32) ? (NCHUNK - g * 32) : 32;
        for (int kcl = 0; kcl < cend; ++kcl) {
            const int kc = g * 32 + kcl;
            ABFrag a0, a1, bf;
            // B fragment: lane's N column = lane&15; element s = K kc*32 + (s>>3)*16 + half*8 + (s&7).
            {
                const __bf16* lp = lw + (kcl * 32 + half * 8) * 16 + l16;
                #pragma unroll
                for (int s = 0; s < 16; ++s)
                    bf.v[s] = lp[((s >> 3) * 16 + (s & 7)) * 16];
            }
            // A fragments: per run, one 16-byte NHWC global load per M tile.
            #pragma unroll
            for (int r = 0; r < 2; ++r) {
                const int Ks  = kc * 32 + r * 16 + half * 8;   // CPAD-const folds all of this
                const int pos = Ks / CPAD;
                const int c0  = Ks % CPAD;
                const int ky  = pos / 5;
                const int kx  = pos - ky * 5;
                const bool pok = (pos < 25);

                int iy = oy0 * STRIDE + ky - 1, ix = ox0 * STRIDE + kx - 1;
                bf16x8 av;
                if (mv0 && pok && (unsigned)iy < (unsigned)Hin && (unsigned)ix < (unsigned)Hin) {
                    av = *reinterpret_cast<const bf16x8*>(
                        act + (((size_t)b0 * Hin + iy) * Hin + ix) * CPAD + c0);
                } else {
                    #pragma unroll
                    for (int j = 0; j < 8; ++j) av[j] = (__bf16)0.0f;
                }
                a0.h[r] = av;

                iy = oy1 * STRIDE + ky - 1; ix = ox1 * STRIDE + kx - 1;
                if (mv1 && pok && (unsigned)iy < (unsigned)Hin && (unsigned)ix < (unsigned)Hin) {
                    av = *reinterpret_cast<const bf16x8*>(
                        act + (((size_t)b1 * Hin + iy) * Hin + ix) * CPAD + c0);
                } else {
                    #pragma unroll
                    for (int j = 0; j < 8; ++j) av[j] = (__bf16)0.0f;
                }
                a1.h[r] = av;
            }
            acc0 = __builtin_amdgcn_wmma_f32_16x16x32_bf16(
                false, a0.v, false, bf.v, (short)0, acc0, false, false);
            acc1 = __builtin_amdgcn_wmma_f32_16x16x32_bf16(
                false, a1.v, false, bf.v, (short)0, acc1, false, false);
        }
    }

    // Store: C/D layout M = r + half*8, N = lane&15.
    const int n    = tn * 16 + l16;
    const float bn = bias[n];
    #pragma unroll
    for (int r = 0; r < 8; ++r) {
        int mg = tm0 * 16 + r + half * 8;
        if (mg < Mtot) out[(size_t)mg * CO + n] = acc0[r] + bn;
        mg += 16;
        if (mg < Mtot) out[(size_t)mg * CO + n] = acc1[r] + bn;
    }
}

// ---------------------------------------------------------------------------
// Kernel 4: per-channel mean/var over M -> scale/shift (y = x*s + t)
// ---------------------------------------------------------------------------
__global__ void ie_bn_stats(const float* __restrict__ x,
                            const float* __restrict__ g,
                            const float* __restrict__ be,
                            float* __restrict__ st, int M, int co)
{
    const int c = blockIdx.x;
    float s = 0.0f, q = 0.0f;
    for (int m = threadIdx.x; m < M; m += blockDim.x) {
        float v = x[(size_t)m * co + c];
        s += v; q += v * v;
    }
    __shared__ float ss[256], qq[256];
    ss[threadIdx.x] = s; qq[threadIdx.x] = q;
    __syncthreads();
    for (int o = 128; o > 0; o >>= 1) {
        if (threadIdx.x < o) { ss[threadIdx.x] += ss[threadIdx.x + o];
                               qq[threadIdx.x] += qq[threadIdx.x + o]; }
        __syncthreads();
    }
    if (threadIdx.x == 0) {
        float inv  = 1.0f / (float)M;
        float mean = ss[0] * inv;
        float var  = qq[0] * inv - mean * mean;
        float sc   = g[c] * rsqrtf(var + 1e-5f);
        st[2 * c]     = sc;
        st[2 * c + 1] = be[c] - mean * sc;
    }
}

// ---------------------------------------------------------------------------
// Kernel 5: normalize + ReLU; restore bf16 NHWC act for next layer;
// optional fp32 NHWC feature copy for pooling.
// ---------------------------------------------------------------------------
__global__ void ie_bn_relu(const float* __restrict__ x,
                           const float* __restrict__ st,
                           __bf16* __restrict__ actout,
                           float*  __restrict__ feat,
                           int total, int co)
{
    int i = blockIdx.x * blockDim.x + threadIdx.x;
    if (i >= total) return;
    int c = i % co;
    float y = fmaxf(x[i] * st[2 * c] + st[2 * c + 1], 0.0f);
    actout[i] = (__bf16)y;
    if (feat) feat[i] = y;
}

// ---------------------------------------------------------------------------
// Kernel 6: vertex projection + bilinear pooling over 4 NHWC feature maps.
// One wave per vertex; lanes stride channels.
// ---------------------------------------------------------------------------
__global__ void __launch_bounds__(128)
ie_pooling(const float* __restrict__ verts,
           const float* __restrict__ f0, const float* __restrict__ f1,
           const float* __restrict__ f2, const float* __restrict__ f3,
           float* __restrict__ out)
{
    const int wave = threadIdx.x >> 5;
    const int lane = threadIdx.x & 31;
    const long v = (long)blockIdx.x * 4 + wave;          // over 32*4096 vertices
    if (v >= 32 * 4096) return;
    const int b = (int)(v >> 12);

    const float* vp = verts + v * 3;
    const float vx = vp[0], vy = vp[1], vz = vp[2];
    // MATRIX = K.dot(RT), hardcoded
    const float p0 =  90.5088f   * vx - 221.7025f * vy -  90.5088f   * vz + 144.8192f;
    const float p1 = -66.257034f * vx                  - 247.274638f * vz + 238.887571f;
    const float p2 =   0.7071f   * vx                  -   0.7071f   * vz +   1.1314f;
    const float inv = 1.0f / (p2 * 256.0f);
    const float xs = p1 * inv;
    const float ys = p0 * inv;

    float* op = out + v * 480;
    const float* fp[4]  = { f0, f1, f2, f3 };
    const int dims[4]   = { 57, 24, 7, 3 };
    const int chans[4]  = { 32, 64, 128, 256 };

    int choff = 0;
    #pragma unroll
    for (int k = 0; k < 4; ++k) {
        const int dim = dims[k], co = chans[k];
        const float fd = (float)dim;
        float cx = fminf(fmaxf(xs * fd, 0.0f), fd - 1.0f);
        float cy = fminf(fmaxf(ys * fd, 0.0f), fd - 1.0f);
        float x1 = floorf(cx), y1 = floorf(cy);
        float x2 = ceilf(cx),  y2 = ceilf(cy);
        float A  = x2 - cx, Bw = cx - x1;
        float G  = y2 - cy, Hh = cy - y1;
        int x1i = (int)x1, y1i = (int)y1, x2i = (int)x2, y2i = (int)y2;
        const float* base = fp[k] + (size_t)b * dim * dim * co;
        const float* pC = base + ((size_t)x1i * dim + y1i) * co;
        const float* pD = base + ((size_t)x1i * dim + y2i) * co;
        const float* pE = base + ((size_t)x2i * dim + y1i) * co;
        const float* pF = base + ((size_t)x2i * dim + y2i) * co;
        const float wC = A * G, wD = Hh * A, wE = G * Bw, wF = Bw * Hh;
        for (int c = lane; c < co; c += 32)
            op[choff + c] = wC * pC[c] + wD * pD[c] + wE * pE[c] + wF * pF[c];
        choff += co;
    }
}

// ---------------------------------------------------------------------------
// Host side
// ---------------------------------------------------------------------------
struct IeLayer { int ci, co, stride, Hin, Hout, featIdx; };

static inline void launch_conv(int cpad, int stride, int co, dim3 grid, hipStream_t st,
                               const __bf16* act, const __bf16* wp, const float* bias,
                               float* out, int M, int Hin, int Hout)
{
    #define IE_CASE(CP, ST, C)                                                    \
        if (cpad == CP && stride == ST && co == C) {                              \
            ie_conv_wmma<CP, ST, C><<<grid, 128, 0, st>>>(act, wp, bias, out,     \
                                                          M, Hin, Hout);          \
            return;                                                               \
        }
    IE_CASE(16, 2, 16)  IE_CASE(16, 1, 16)  IE_CASE(16, 2, 32)
    IE_CASE(32, 1, 32)  IE_CASE(32, 2, 64)  IE_CASE(64, 1, 64)
    IE_CASE(64, 2, 128) IE_CASE(128, 1, 128) IE_CASE(128, 2, 256)
    #undef IE_CASE
}

extern "C" void kernel_launch(void* const* d_in, const int* in_sizes, int n_in,
                              void* d_out, int out_size, void* d_ws, size_t ws_size,
                              hipStream_t stream)
{
    (void)in_sizes; (void)n_in; (void)out_size; (void)ws_size;

    static const IeLayer L[13] = {
        {  6,  16, 2, 256, 127, -1}, { 16,  16, 1, 127, 125, -1}, { 16,  16, 1, 125, 123, -1},
        { 16,  32, 2, 123,  61, -1}, { 32,  32, 1,  61,  59, -1}, { 32,  32, 1,  59,  57,  0},
        { 32,  64, 2,  57,  28, -1}, { 64,  64, 1,  28,  26, -1}, { 64,  64, 1,  26,  24,  1},
        { 64, 128, 2,  24,  11, -1}, {128, 128, 1,  11,   9, -1}, {128, 128, 1,   9,   7,  2},
        {128, 256, 2,   7,   3,  3},
    };

    char* ws = (char*)d_ws;
    const float* img_occ   = (const float*)d_in[0];
    const float* img_unocc = (const float*)d_in[1];
    const float* verts     = (const float*)d_in[2];

    __bf16* wprep = (__bf16*)(ws + OFF_WPREP);
    float*  conv  = (float*)(ws + OFF_CONV);
    float*  stats = (float*)(ws + OFF_STATS);
    const size_t featOff[4] = { OFF_FEAT0, OFF_FEAT1, OFF_FEAT2, OFF_FEAT3 };

    // Layer 0 input: concat + pad to 16 channels, NHWC bf16.
    ie_concat_pad<<<(32 * 65536 + 255) / 256, 256, 0, stream>>>(
        img_occ, img_unocc, (__bf16*)(ws + OFF_ACT0));

    size_t actOff[2] = { OFF_ACT0, OFF_ACT1 };
    for (int l = 0; l < 13; ++l) {
        const int ci = L[l].ci, co = L[l].co, s = L[l].stride;
        const int Hin = L[l].Hin, Hout = L[l].Hout;
        const int cpad = (ci < 16) ? 16 : ci;
        int lgc = 0; while ((1 << lgc) < cpad) ++lgc;
        const int nchunk = (25 * cpad + 31) / 32;
        const int Ktot   = nchunk * 32;
        const int M      = 32 * Hout * Hout;

        const float* wl  = (const float*)d_in[3 + 4 * l + 0];
        const float* bl  = (const float*)d_in[3 + 4 * l + 1];
        const float* gl  = (const float*)d_in[3 + 4 * l + 2];
        const float* bel = (const float*)d_in[3 + 4 * l + 3];

        __bf16* actin  = (__bf16*)(ws + actOff[l & 1]);
        __bf16* actout = (__bf16*)(ws + actOff[(l + 1) & 1]);

        ie_prep_weights<<<(Ktot * co + 255) / 256, 256, 0, stream>>>(
            wl, wprep, ci, co, lgc, Ktot);

        // 4 waves/block, 2 M-tiles per wave -> 8 M-tiles (128 rows) per block.
        dim3 cgrid(((M + 15) / 16 + 7) / 8, co / 16);
        launch_conv(cpad, s, co, cgrid, stream, actin, wprep, bl, conv, M, Hin, Hout);

        ie_bn_stats<<<co, 256, 0, stream>>>(conv, gl, bel, stats, M, co);

        float* feat = (L[l].featIdx >= 0) ? (float*)(ws + featOff[L[l].featIdx]) : nullptr;
        const long total = (long)M * co;
        ie_bn_relu<<<(unsigned)((total + 255) / 256), 256, 0, stream>>>(
            conv, stats, actout, feat, (int)total, co);
    }

    ie_pooling<<<(32 * 4096) / 4, 128, 0, stream>>>(
        verts,
        (const float*)(ws + OFF_FEAT0), (const float*)(ws + OFF_FEAT1),
        (const float*)(ws + OFF_FEAT2), (const float*)(ws + OFF_FEAT3),
        (float*)d_out);
}
